// MultiheadLinearAttention_29910152249748
// MI455X (gfx1250) — matrix operand order repaired and checked by reading, not compiled, not run on hardware
//
#include <hip/hip_runtime.h>
#include <hip/hip_bf16.h>

typedef __attribute__((ext_vector_type(16))) __bf16 v16bf;
typedef __attribute__((ext_vector_type(8)))  __bf16 v8bf;
typedef __attribute__((ext_vector_type(4)))  __bf16 v4bf;
typedef __attribute__((ext_vector_type(8)))  float  v8f;

#define B_  4
#define T_  8192
#define D_  1024
#define H_  16
#define DH_ 64
#define M_  (B_ * T_)    // 32768 rows
#define TSPLIT 8         // T-chunks for KV / Ksum reduction

// ---------------------------------------------------------------------------
// WMMA bf16 fragment helpers (wave32, 16x16x32, per CDNA5 ISA layouts)
// A (16x32, MxK): lane 0-15 -> M=lane, K = {k0+0..7, k0+16..23}
//                 lane 16-31 -> M=lane-16, K = {k0+8..15, k0+24..31}
// B (32x16, KxN): lane 0-15 -> N=lane, K = {k0+0..7, k0+16..23}; lanes 16-31 -> K = {k0+8..15, k0+24..31}
// C (16x16 f32):  VGPR r: lanes 0-15 -> (M=r, N=lane); lanes 16-31 -> (M=r+8)
// ---------------------------------------------------------------------------
__device__ __forceinline__ v16bf frag_a(const __bf16* rowA, int k0, int hl) {
  const __bf16* p = rowA + k0 + 8 * hl;
  v8bf lo = *(const v8bf*)(p);
  v8bf hi = *(const v8bf*)(p + 16);
  v16bf a;
#pragma unroll
  for (int i = 0; i < 8; ++i) { a[i] = lo[i]; a[i + 8] = hi[i]; }
  return a;
}

__device__ __forceinline__ v16bf frag_b(const __bf16* rowB, int k0, int hl) {
  const __bf16* p = rowB + k0 + 8 * hl;
  v8bf lo = *(const v8bf*)(p);
  v8bf hi = *(const v8bf*)(p + 16);
  v16bf b;
#pragma unroll
  for (int i = 0; i < 8; ++i) { b[i] = lo[i]; b[i + 8] = hi[i]; }
  return b;
}

__device__ __forceinline__ v8f wmma_bf16(v16bf a, v16bf b, v8f c) {
  return __builtin_amdgcn_wmma_f32_16x16x32_bf16(
      false, a, false, b, (short)0, c, false, false);
}

__device__ __forceinline__ float phi_fn(float z) {
  // elu(z / sqrt(Dh)) + 1, sqrt(64) = 8
  float s = z * 0.125f;
  return (s > 0.0f) ? (s + 1.0f) : __expf(s);
}

// ---------------------------------------------------------------------------
// fp32 -> bf16 conversion (float4 in, 4x bf16 (8B) out)
// ---------------------------------------------------------------------------
__global__ void k_cvt_bf16(const float* __restrict__ src, __bf16* __restrict__ dst,
                           size_t n) {
  size_t i = ((size_t)blockIdx.x * blockDim.x + threadIdx.x) * 4;
  size_t stride = (size_t)gridDim.x * blockDim.x * 4;
  for (; i < n; i += stride) {
    float4 f = *(const float4*)(src + i);
    v4bf o;
    o[0] = (__bf16)f.x; o[1] = (__bf16)f.y; o[2] = (__bf16)f.z; o[3] = (__bf16)f.w;
    *(v4bf*)(dst + i) = o;
  }
}

__global__ void k_zero(float* __restrict__ p, size_t n) {
  size_t i = (size_t)blockIdx.x * blockDim.x + threadIdx.x;
  size_t stride = (size_t)gridDim.x * blockDim.x;
  for (; i < n; i += stride) p[i] = 0.0f;
}

// ---------------------------------------------------------------------------
// Projection epilogue for one 16x16 C tile.
// MODE 0: Q -> phi * mask, layout [B,H,T,Dh] (strided scalar stores)
// MODE 1: K -> phi * mask, layout [B,H,Dh,T] (one packed v8bf store per lane)
// MODE 2: V ->        mask, layout [B,H,Dh,T] (one packed v8bf store per lane)
// ---------------------------------------------------------------------------
template <int MODE>
__device__ __forceinline__ void proj_epilogue(v8f acc, int m0, int n, int hl,
                                              const float* __restrict__ bias,
                                              const int* __restrict__ mask,
                                              __bf16* __restrict__ out) {
  const int h = n >> 6;
  const int d = n & (DH_ - 1);
  const float bn = bias[n];
  if (MODE == 0) {
#pragma unroll
    for (int r = 0; r < 8; ++r) {
      const int m = m0 + r + 8 * hl;
      const int b = m >> 13;
      const int t = m & (T_ - 1);
      const float mk = (float)mask[(size_t)b * T_ + t];
      float v = phi_fn(acc[r] + bn) * mk;
      out[((size_t)(b * H_ + h) * T_ + t) * DH_ + d] = (__bf16)v;
    }
  } else {
    const int mb = m0 + 8 * hl;            // first of 8 consecutive rows
    const int b = mb >> 13;
    const int t0 = mb & (T_ - 1);
    v8bf pk;
#pragma unroll
    for (int r = 0; r < 8; ++r) {
      const float mk = (float)mask[(size_t)b * T_ + t0 + r];
      float v = acc[r] + bn;
      if (MODE == 1) v = phi_fn(v);
      pk[r] = (__bf16)(v * mk);
    }
    *(v8bf*)(out + ((size_t)(b * H_ + h) * DH_ + d) * T_ + t0) = pk;
  }
}

// ---------------------------------------------------------------------------
// Projection GEMM: y[m,n] = sum_k xbf[m,k] * W[n,k] + bias[n]
// 4x2 register blocking: each wave computes 64x32 of C (8 accumulators);
// 12 b128 loads feed 8 wmma per 32-K step (B frags reused 4x, A frags 2x).
// Block = 128 threads = 4 waves over N. Grid: (M/64, D/128).
// ---------------------------------------------------------------------------
template <int MODE>
__global__ __launch_bounds__(128)
void k_proj(const __bf16* __restrict__ xbf, const __bf16* __restrict__ W,
            const float* __restrict__ bias, const int* __restrict__ mask,
            __bf16* __restrict__ out) {
  const int lane = threadIdx.x & 31;
  const int wave = threadIdx.x >> 5;
  const int ml = lane & 15;
  const int hl = lane >> 4;
  const int m0 = blockIdx.x * 64;
  const int n0 = blockIdx.y * 128 + wave * 32;

  const __bf16* rowA = xbf + (size_t)(m0 + ml) * D_;
  const __bf16* rowB0 = W + (size_t)(n0 + ml) * D_;
  const __bf16* rowB1 = rowB0 + (size_t)16 * D_;

  v8f acc[4][2];
#pragma unroll
  for (int mi = 0; mi < 4; ++mi)
#pragma unroll
    for (int ni = 0; ni < 2; ++ni) acc[mi][ni] = (v8f){};

  for (int k0 = 0; k0 < D_; k0 += 32) {
    v16bf b0 = frag_b(rowB0, k0, hl);
    v16bf b1 = frag_b(rowB1, k0, hl);
#pragma unroll
    for (int mi = 0; mi < 4; ++mi) {
      v16bf a = frag_a(rowA + (size_t)(mi * 16) * D_, k0, hl);
      acc[mi][0] = wmma_bf16(a, b0, acc[mi][0]);
      acc[mi][1] = wmma_bf16(a, b1, acc[mi][1]);
    }
  }

#pragma unroll
  for (int mi = 0; mi < 4; ++mi) {
    proj_epilogue<MODE>(acc[mi][0], m0 + mi * 16, n0 + ml,      hl, bias, mask, out);
    proj_epilogue<MODE>(acc[mi][1], m0 + mi * 16, n0 + 16 + ml, hl, bias, mask, out);
  }
}

// ---------------------------------------------------------------------------
// Ksum[bh,d] = sum_t k_t[bh,d,t], T split into TSPLIT chunks + atomic combine.
// Grid: (B*H, TSPLIT) x 64 threads.
// ---------------------------------------------------------------------------
__global__ void k_ksum(const __bf16* __restrict__ k_t, float* __restrict__ ksum) {
  const int bh = blockIdx.x;
  const int chunk = blockIdx.y;
  const int d = threadIdx.x;
  const __bf16* p = k_t + ((size_t)bh * DH_ + d) * T_ + chunk * (T_ / TSPLIT);
  float s = 0.0f;
  for (int t = 0; t < T_ / TSPLIT; t += 8) {
    v8bf v = *(const v8bf*)(p + t);
#pragma unroll
    for (int i = 0; i < 8; ++i) s += (float)v[i];
  }
  atomicAdd(&ksum[bh * DH_ + d], s);
}

// ---------------------------------------------------------------------------
// KV[bh, f, d] = sum_t v[t,f] * k[t,d]  (= v^T k), WMMA over a T-chunk,
// partials combined with global_atomic_add_f32 into fp32 kvf.
// Block = 128 threads = 4 waves, each wave a 32x32 tile (2x2 blocking).
// Grid: (B*H, TSPLIT).
// ---------------------------------------------------------------------------
__global__ __launch_bounds__(128)
void k_kv(const __bf16* __restrict__ v_t, const __bf16* __restrict__ k_t,
          float* __restrict__ kvf) {
  const int bh = blockIdx.x;
  const int kbase = blockIdx.y * (T_ / TSPLIT);
  const int lane = threadIdx.x & 31;
  const int wave = threadIdx.x >> 5;   // 0..3
  const int ml = lane & 15;
  const int hl = lane >> 4;
  const int tm = (wave >> 1) * 32;     // f-offset
  const int tn = (wave & 1) * 32;      // d-offset

  const __bf16* rowA0 = v_t + ((size_t)bh * DH_ + tm + ml) * T_ + kbase;
  const __bf16* rowA1 = rowA0 + (size_t)16 * T_;
  const __bf16* rowB0 = k_t + ((size_t)bh * DH_ + tn + ml) * T_ + kbase;
  const __bf16* rowB1 = rowB0 + (size_t)16 * T_;

  v8f acc00 = {}, acc01 = {}, acc10 = {}, acc11 = {};
#pragma unroll 2
  for (int k0 = 0; k0 < T_ / TSPLIT; k0 += 32) {
    v16bf a0 = frag_a(rowA0, k0, hl);
    v16bf a1 = frag_a(rowA1, k0, hl);
    v16bf b0 = frag_b(rowB0, k0, hl);
    v16bf b1 = frag_b(rowB1, k0, hl);
    acc00 = wmma_bf16(a0, b0, acc00);
    acc01 = wmma_bf16(a0, b1, acc01);
    acc10 = wmma_bf16(a1, b0, acc10);
    acc11 = wmma_bf16(a1, b1, acc11);
  }

  float* outp = kvf + (size_t)bh * DH_ * DH_;
#pragma unroll
  for (int r = 0; r < 8; ++r) {
    const int r0 = tm + r + 8 * hl;
    const int r1 = tm + 16 + r + 8 * hl;
    atomicAdd(&outp[r0 * DH_ + tn + ml],      acc00[r]);
    atomicAdd(&outp[r0 * DH_ + tn + 16 + ml], acc01[r]);
    atomicAdd(&outp[r1 * DH_ + tn + ml],      acc10[r]);
    atomicAdd(&outp[r1 * DH_ + tn + 16 + ml], acc11[r]);
  }
}

// ---------------------------------------------------------------------------
// den[bh, t] = sum_d q[bh,t,d] * Ksum[bh,d]
// ---------------------------------------------------------------------------
__global__ void k_den(const __bf16* __restrict__ q, const float* __restrict__ ksum,
                      float* __restrict__ den) {
  size_t idx = (size_t)blockIdx.x * blockDim.x + threadIdx.x;  // < B*H*T
  const size_t bh = idx / T_;
  const __bf16* qp = q + idx * DH_;
  const float* ks = ksum + bh * DH_;
  float s = 0.0f;
#pragma unroll
  for (int d0 = 0; d0 < DH_; d0 += 8) {
    v8bf v = *(const v8bf*)(qp + d0);
#pragma unroll
    for (int i = 0; i < 8; ++i) s += (float)v[i] * ks[d0 + i];
  }
  den[idx] = s;
}

// ---------------------------------------------------------------------------
// num[t,f] = sum_d q[t,d] * KVt[f,d]; out = num/(den+eps) -> bf16 [B,T,D]
// Each wave: one 16-row t-tile x all 4 f-tiles (A frags reused 4x).
// Grid: (T/64, B*H), block 128.
// ---------------------------------------------------------------------------
__global__ __launch_bounds__(128)
void k_attn(const __bf16* __restrict__ q, const __bf16* __restrict__ kvt,
            const float* __restrict__ den, __bf16* __restrict__ attn) {
  const int bh = blockIdx.y;
  const int b = bh >> 4;               // H = 16
  const int h = bh & (H_ - 1);
  const int lane = threadIdx.x & 31;
  const int wave = threadIdx.x >> 5;   // 0..3
  const int ml = lane & 15;
  const int hl = lane >> 4;
  const int t0 = blockIdx.x * 64 + wave * 16;

  const __bf16* rowA = q + ((size_t)bh * T_ + t0 + ml) * DH_;
  v16bf a0 = frag_a(rowA, 0, hl);
  v16bf a1 = frag_a(rowA, 32, hl);

  const __bf16* Bbase = kvt + (size_t)bh * DH_ * DH_;
#pragma unroll
  for (int nt = 0; nt < 4; ++nt) {
    const __bf16* rowB = Bbase + (size_t)(nt * 16 + ml) * DH_;
    v8f acc = {};
    acc = wmma_bf16(a0, frag_b(rowB, 0, hl), acc);
    acc = wmma_bf16(a1, frag_b(rowB, 32, hl), acc);
    const int f = nt * 16 + ml;
#pragma unroll
    for (int r = 0; r < 8; ++r) {
      const int t = t0 + r + 8 * hl;
      const float dv = den[(size_t)bh * T_ + t] + 1e-6f;
      attn[((size_t)b * T_ + t) * D_ + h * DH_ + f] = (__bf16)(acc[r] / dv);
    }
  }
}

// ---------------------------------------------------------------------------
// Output projection: out[m,n] = sum_k attn[m,k] * Wo[n,k] + bo[n]  (fp32 out)
// Same 4x2 register blocking as k_proj.
// ---------------------------------------------------------------------------
__global__ __launch_bounds__(128)
void k_outproj(const __bf16* __restrict__ attn, const __bf16* __restrict__ W,
               const float* __restrict__ bias, float* __restrict__ out) {
  const int lane = threadIdx.x & 31;
  const int wave = threadIdx.x >> 5;
  const int ml = lane & 15;
  const int hl = lane >> 4;
  const int m0 = blockIdx.x * 64;
  const int n0 = blockIdx.y * 128 + wave * 32;

  const __bf16* rowA = attn + (size_t)(m0 + ml) * D_;
  const __bf16* rowB0 = W + (size_t)(n0 + ml) * D_;
  const __bf16* rowB1 = rowB0 + (size_t)16 * D_;

  v8f acc[4][2];
#pragma unroll
  for (int mi = 0; mi < 4; ++mi)
#pragma unroll
    for (int ni = 0; ni < 2; ++ni) acc[mi][ni] = (v8f){};

  for (int k0 = 0; k0 < D_; k0 += 32) {
    v16bf b0 = frag_b(rowB0, k0, hl);
    v16bf b1 = frag_b(rowB1, k0, hl);
#pragma unroll
    for (int mi = 0; mi < 4; ++mi) {
      v16bf a = frag_a(rowA + (size_t)(mi * 16) * D_, k0, hl);
      acc[mi][0] = wmma_bf16(a, b0, acc[mi][0]);
      acc[mi][1] = wmma_bf16(a, b1, acc[mi][1]);
    }
  }

  const float bn0 = bias[n0 + ml];
  const float bn1 = bias[n0 + 16 + ml];
#pragma unroll
  for (int mi = 0; mi < 4; ++mi) {
#pragma unroll
    for (int r = 0; r < 8; ++r) {
      const int m = m0 + mi * 16 + r + 8 * hl;
      out[(size_t)m * D_ + n0 + ml]      = acc[mi][0][r] + bn0;
      out[(size_t)m * D_ + n0 + 16 + ml] = acc[mi][1][r] + bn1;
    }
  }
}

// ---------------------------------------------------------------------------
// Host launcher
// ---------------------------------------------------------------------------
extern "C" void kernel_launch(void* const* d_in, const int* in_sizes, int n_in,
                              void* d_out, int out_size, void* d_ws, size_t ws_size,
                              hipStream_t stream) {
  const float* x    = (const float*)d_in[0];
  const int*   mask = (const int*)d_in[1];
  const float* Wq   = (const float*)d_in[2];
  const float* bq   = (const float*)d_in[3];
  const float* Wk   = (const float*)d_in[4];
  const float* bk   = (const float*)d_in[5];
  const float* Wv   = (const float*)d_in[6];
  const float* bv   = (const float*)d_in[7];
  const float* Wo   = (const float*)d_in[8];
  const float* bo   = (const float*)d_in[9];
  float* out = (float*)d_out;

  char* ws = (char*)d_ws;
  size_t off = 0;
  auto alloc = [&](size_t bytes) -> void* {
    void* p = ws + off;
    off += (bytes + 255) & ~(size_t)255;
    return p;
  };

  const size_t NXD = (size_t)M_ * D_;              // 33,554,432 elements
  const size_t NKV = (size_t)B_ * H_ * DH_ * DH_;  // 262,144
  const size_t NKS = (size_t)B_ * H_ * DH_;        // 4,096
  __bf16* xbf  = (__bf16*)alloc(NXD * 2);
  __bf16* wqb  = (__bf16*)alloc((size_t)D_ * D_ * 2);
  __bf16* wkb  = (__bf16*)alloc((size_t)D_ * D_ * 2);
  __bf16* wvb  = (__bf16*)alloc((size_t)D_ * D_ * 2);
  __bf16* wob  = (__bf16*)alloc((size_t)D_ * D_ * 2);
  __bf16* qb   = (__bf16*)alloc(NXD * 2);   // [B,H,T,Dh]
  __bf16* ktb  = (__bf16*)alloc(NXD * 2);   // [B,H,Dh,T]
  __bf16* vtb  = (__bf16*)alloc(NXD * 2);   // [B,H,Dh,T]
  float*  kvf  = (float*)alloc(NKV * 4);    // fp32 KV accumulator (atomics)
  float*  ksum = (float*)alloc(NKS * 4);    // fp32 Ksum accumulator (atomics)
  __bf16* kvtb = (__bf16*)alloc(NKV * 2);   // KVt bf16 (B operand for attn)
  float*  den  = (float*)alloc((size_t)B_ * H_ * T_ * 4);
  __bf16* attn = (__bf16*)alloc(NXD * 2);   // [B,T,D]
  (void)ws_size; (void)in_sizes; (void)n_in; (void)out_size;

  // 0) zero the atomic accumulators (kvf and ksum are adjacent, 256B-aligned)
  k_zero<<<512, 256, 0, stream>>>(kvf, NKV + NKS);

  // 1) fp32 -> bf16 staging
  k_cvt_bf16<<<4096, 256, 0, stream>>>(x, xbf, NXD);
  k_cvt_bf16<<<512, 256, 0, stream>>>(Wq, wqb, (size_t)D_ * D_);
  k_cvt_bf16<<<512, 256, 0, stream>>>(Wk, wkb, (size_t)D_ * D_);
  k_cvt_bf16<<<512, 256, 0, stream>>>(Wv, wvb, (size_t)D_ * D_);
  k_cvt_bf16<<<512, 256, 0, stream>>>(Wo, wob, (size_t)D_ * D_);

  // 2) Q/K/V projections (WMMA, 4x2 blocked) with fused phi+mask+transpose
  dim3 gproj(M_ / 64, D_ / 128);   // 512 x 8
  k_proj<0><<<gproj, 128, 0, stream>>>(xbf, wqb, bq, mask, qb);
  k_proj<1><<<gproj, 128, 0, stream>>>(xbf, wkb, bk, mask, ktb);
  k_proj<2><<<gproj, 128, 0, stream>>>(xbf, wvb, bv, mask, vtb);

  // 3) Ksum + KV = v^T k (WMMA over T, split-T, atomic combine), then ->bf16
  dim3 gred(B_ * H_, TSPLIT);      // 64 x 8
  k_ksum<<<gred, DH_, 0, stream>>>(ktb, ksum);
  k_kv<<<gred, 128, 0, stream>>>(vtb, ktb, kvf);
  k_cvt_bf16<<<256, 256, 0, stream>>>(kvf, kvtb, NKV);

  // 4) den = q . Ksum ; num = q KV^T (WMMA) with fused normalize
  k_den<<<(B_ * H_ * T_) / 256, 256, 0, stream>>>(qb, ksum, den);
  dim3 gattn(T_ / 64, B_ * H_);    // 128 x 64
  k_attn<<<gattn, 128, 0, stream>>>(qb, kvtb, den, attn);

  // 5) Output projection (WMMA, 4x2 blocked), fp32 result
  k_outproj<<<gproj, 128, 0, stream>>>(attn, wob, bo, out);
}
